// ModalityAdaptiveModule_31224412242035
// MI455X (gfx1250) — matrix operand order repaired and checked by reading, not compiled, not run on hardware
//
#include <hip/hip_runtime.h>

// ---------------------------------------------------------------------------
// MI455X / gfx1250: f16 WMMA (16x16x32) everywhere, flash-attention streaming
// with TDM (tensor_load_to_lds) double-buffered K/V staging in LDS.
// ---------------------------------------------------------------------------

typedef _Float16 half8  __attribute__((ext_vector_type(8)));
typedef _Float16 half16 __attribute__((ext_vector_type(16)));
typedef float    f32x8  __attribute__((ext_vector_type(8)));
typedef float    f32x4  __attribute__((ext_vector_type(4)));
typedef unsigned int u32x4 __attribute__((ext_vector_type(4)));
typedef int          i32x4 __attribute__((ext_vector_type(4)));
typedef int          i32x8 __attribute__((ext_vector_type(8)));

#define SEQ 4096
#define DIMV 512
#define NB  4    // combined batches (2 text + 2 img)
#define KT  32   // keys per flash-attention stage

union HVec { half16 v; half8 h[2]; };

// A-fragment (16x32 f16, MxK), CDNA5 layout: lane L holds row M=L&15;
// halves [kb..kb+7] and [kb+16..kb+23], kb=(L>>4)*8.
__device__ __forceinline__ half16 load_afrag(const _Float16* __restrict__ tile, int stride) {
  int lane = threadIdx.x & 31;
  const _Float16* p = tile + (size_t)(lane & 15) * stride + ((lane >> 4) << 3);
  HVec u;
  u.h[0] = *(const half8*)(p);
  u.h[1] = *(const half8*)(p + 16);
  return u.v;
}

// B-fragment (32x16 f16, KxN) from row-major src[n][k]: lane L holds column
// N=L&15; halves [kb..kb+7] and [kb+16..kb+23], kb=(L>>4)*8.
__device__ __forceinline__ half16 load_bfrag(const _Float16* __restrict__ tile, int stride) {
  int lane = threadIdx.x & 31;
  const _Float16* p = tile + (size_t)(lane & 15) * stride + ((lane >> 4) << 3);
  HVec u;
  u.h[0] = *(const half8*)(p);
  u.h[1] = *(const half8*)(p + 16);
  return u.v;
}

__device__ __forceinline__ f32x8 wmma16x16x32(half16 a, half16 b, f32x8 c) {
  return __builtin_amdgcn_wmma_f32_16x16x32_f16(false, a, false, b, (short)0, c, false, false);
}

// Generic shared pointer -> LDS byte offset (aperture high bits live in
// addr[63:32]; low 32 bits are the in-allocation offset).
__device__ __forceinline__ unsigned lds_off(const void* p) {
  return (unsigned)(unsigned long long)p;
}

// ---------------------------------------------------------------------------
// TDM 2D descriptor (ISA ch.8): D# group0/group1, groups 2-3 zero (<=2D).
// data_size = 2 bytes. Optional LDS padding: interval code -> 2^(c+1) dwords,
// amount code -> (c+1) dwords inserted.
// This toolchain's builtin takes 6 args (g0, g1, g2, g3, g4, cpol).
// ---------------------------------------------------------------------------
__device__ __forceinline__ void tdm_load_2d(
    unsigned ldsByteOff, const void* gaddr,
    unsigned tensorD0, unsigned tensorD1,
    unsigned tileD0, unsigned tileD1,
    unsigned strideD0,
    unsigned padIntervalCode, unsigned padAmountCode, bool padEn) {
  unsigned long long ga = (unsigned long long)gaddr;
  u32x4 g0;
  g0[0] = 1u;                                        // count=1, user descriptor
  g0[1] = ldsByteOff;                                // lds_addr
  g0[2] = (unsigned)(ga & 0xffffffffu);              // global_addr[31:0]
  g0[3] = (unsigned)((ga >> 32) & 0x01ffffffu) | (2u << 30); // addr[56:32], type=2
  unsigned w0 = (1u << 16);                          // data_size = 2B
  if (padEn) w0 |= (1u << 20) | (padIntervalCode << 22) | (padAmountCode << 25);
  i32x8 g1;
  g1[0] = (int)w0;                                   // mask=0, flags
  g1[1] = (int)((tensorD0 & 0xffffu) << 16);         // dim0[15:0] @ bits63:48
  g1[2] = (int)(((tensorD0 >> 16) & 0xffffu) | ((tensorD1 & 0xffffu) << 16));
  g1[3] = (int)(((tensorD1 >> 16) & 0xffffu) | ((tileD0 & 0xffffu) << 16));
  g1[4] = (int)(tileD1 & 0xffffu);                   // tile_dim1, tile_dim2=0
  g1[5] = (int)strideD0;                             // dim0_stride[31:0]
  g1[6] = 0;                                         // stride hi, dim1_stride lo
  g1[7] = 0;
  i32x4 z4 = {0, 0, 0, 0};
  i32x8 z8 = {0, 0, 0, 0, 0, 0, 0, 0};
  __builtin_amdgcn_tensor_load_to_lds(g0, g1, z4, z4, z8, 0);
}

// ---------------------------------------------------------------------------
// Kernel 0: weights fp32 -> f16; Wq/bq pre-scaled by D^-0.5 * log2(e).
// WH layout: [Wq, Wkt, Wvt, Wki, Wvi, Wo], each 512*512.
// ---------------------------------------------------------------------------
__global__ __launch_bounds__(256) void wcvt_kernel(
    const float* __restrict__ Wq, const float* __restrict__ Wkt,
    const float* __restrict__ Wvt, const float* __restrict__ Wki,
    const float* __restrict__ Wvi, const float* __restrict__ Wo,
    const float* __restrict__ bq, _Float16* __restrict__ WH,
    float* __restrict__ bqS, float qscale) {
  int i = blockIdx.x * 256 + threadIdx.x;
  int m = i >> 18;
  int off = i & 262143;
  const float* src;
  float sc = 1.f;
  switch (m) {
    case 0: src = Wq;  sc = qscale; break;
    case 1: src = Wkt; break;
    case 2: src = Wvt; break;
    case 3: src = Wki; break;
    case 4: src = Wvi; break;
    default: src = Wo; break;
  }
  WH[i] = (_Float16)(src[off] * sc);
  if (i < DIMV) bqS[i] = bq[i] * qscale;
}

// ---------------------------------------------------------------------------
// Kernel 1: input LayerNorms -> f16 norms[4][4096][512].
// ---------------------------------------------------------------------------
__global__ __launch_bounds__(256) void ln_in_kernel(
    const float* __restrict__ text, const float* __restrict__ img,
    const float* __restrict__ gt, const float* __restrict__ bt,
    const float* __restrict__ gi, const float* __restrict__ bi,
    _Float16* __restrict__ normsH) {
  int lane = threadIdx.x & 31;
  int wv = threadIdx.x >> 5;
  int row = blockIdx.x * 8 + wv;
  int b = row >> 12;
  int s = row & 4095;
  const float* src = (b < 2) ? (text + ((size_t)b * SEQ + s) * DIMV)
                             : (img + ((size_t)(b - 2) * SEQ + s) * DIMV);
  const float* g  = (b < 2) ? gt : gi;
  const float* be = (b < 2) ? bt : bi;
  float x[16];
#pragma unroll
  for (int j = 0; j < 4; ++j) {
    f32x4 t = *(const f32x4*)(src + lane * 16 + j * 4);
    x[j*4+0]=t[0]; x[j*4+1]=t[1]; x[j*4+2]=t[2]; x[j*4+3]=t[3];
  }
  float sum = 0.f, sq = 0.f;
#pragma unroll
  for (int j = 0; j < 16; ++j) { sum += x[j]; sq += x[j]*x[j]; }
#pragma unroll
  for (int off = 1; off < 32; off <<= 1) {
    sum += __shfl_xor(sum, off);
    sq  += __shfl_xor(sq, off);
  }
  float mu = sum * (1.f/DIMV);
  float var = sq * (1.f/DIMV) - mu*mu;
  float rstd = __builtin_amdgcn_rsqf(var + 1e-5f);
  half8 o[2];
#pragma unroll
  for (int j = 0; j < 16; ++j) {
    int d = lane*16 + j;
    o[j>>3][j&7] = (_Float16)((x[j]-mu)*rstd*g[d] + be[d]);
  }
  _Float16* dst = normsH + (size_t)row*DIMV + lane*16;
  *(half8*)(dst)   = o[0];
  *(half8*)(dst+8) = o[1];
}

// ---------------------------------------------------------------------------
// Kernel 2: Q/K/V projections (one 16x16 tile per wave, 16 WMMA k-steps).
// Q (pre-scaled) and K row-major; V stored transposed (Vt[b][d][s]).
// ---------------------------------------------------------------------------
__global__ __launch_bounds__(256) void proj_kernel(
    const _Float16* __restrict__ normsH, const _Float16* __restrict__ WH,
    const float* __restrict__ bqS, const float* __restrict__ bkt,
    const float* __restrict__ bki, const float* __restrict__ bvt,
    const float* __restrict__ bvi,
    _Float16* __restrict__ QH, _Float16* __restrict__ KH,
    _Float16* __restrict__ VtH) {
  int lane = threadIdx.x & 31;
  int wv = threadIdx.x >> 5;
  int tile = blockIdx.x * 8 + wv;
  int type = tile >> 15;
  int rem  = tile & 32767;
  int m0 = (rem >> 5) << 4;
  int n0 = (rem & 31) << 4;
  int batch = m0 >> 12;
  bool firstHalf = batch < 2;
  const _Float16* W;
  const float* bias;
  if (type == 0)      { W = WH;                                bias = bqS; }
  else if (type == 1) { W = WH + (firstHalf ? 1 : 3) * 262144; bias = firstHalf ? bkt : bki; }
  else                { W = WH + (firstHalf ? 2 : 4) * 262144; bias = firstHalf ? bvt : bvi; }

  const f32x8 zero8 = {0.f,0.f,0.f,0.f,0.f,0.f,0.f,0.f};
  f32x8 acc = zero8;
#pragma unroll
  for (int kk = 0; kk < 16; ++kk) {
    half16 a  = load_afrag(normsH + (size_t)m0 * DIMV + kk*32, DIMV);
    half16 bf = load_bfrag(W      + (size_t)n0 * DIMV + kk*32, DIMV);
    acc = wmma16x16x32(a, bf, acc);
  }
  int rbase = (lane >> 4) << 3;
  int cidx  = lane & 15;
  int n = n0 + cidx;
  float bv = bias[n];
  if (type == 2) {
    half8 ph;
#pragma unroll
    for (int v = 0; v < 8; ++v) ph[v] = (_Float16)(acc[v] + bv);
    int s = (m0 & 4095) + rbase;
    *(half8*)(VtH + ((size_t)batch * DIMV + n) * SEQ + s) = ph;
  } else {
    _Float16* dst = (type == 0) ? QH : KH;
#pragma unroll
    for (int v = 0; v < 8; ++v)
      dst[(size_t)(m0 + rbase + v) * DIMV + n] = (_Float16)(acc[v] + bv);
  }
}

// ---------------------------------------------------------------------------
// Kernel 3: flash attention with TDM-staged K/V.
//   - wave = 16 queries, WG = 128 queries; Q fragments in registers.
//   - per 32-key stage: K tile (32x512, LDS stride 520 halfs) and Vt tile
//     (512x32, LDS stride 40 halfs) DMA'd by wave 0 via tensor_load_to_lds
//     with HW padding -> conflict-free ds_load B-fragments; double buffered,
//     synchronized with s_wait_tensorcnt + workgroup barriers.
//   - softmax in exp2 domain; P re-fragmented through padded per-wave LDS.
// ---------------------------------------------------------------------------
#define KSTRIDE 520   // 512 + 4 dwords pad (interval code 7 = 256 dwords)
#define VSTRIDE 40    // 32 + 4 dwords pad  (interval code 3 = 16 dwords)
#define PSTRIDE 40

__global__ __launch_bounds__(256) void fa_kernel(
    const _Float16* __restrict__ QH, const _Float16* __restrict__ KH,
    const _Float16* __restrict__ VtH, _Float16* __restrict__ XH) {
  __shared__ _Float16 Kb[2][KT * KSTRIDE];     //  65 KB
  __shared__ _Float16 Vb[2][DIMV * VSTRIDE];   //  80 KB
  __shared__ _Float16 pbuf[8][16 * PSTRIDE];   //  10 KB
  int lane = threadIdx.x & 31;
  int wv = threadIdx.x >> 5;
  int b  = blockIdx.x >> 5;
  int qb = blockIdx.x & 31;
  int m0 = qb * 128 + wv * 16;
  const _Float16* Q  = QH  + (size_t)b * SEQ * DIMV;
  const _Float16* K  = KH  + (size_t)b * SEQ * DIMV;
  const _Float16* Vt = VtH + (size_t)b * DIMV * SEQ;
  _Float16* X = XH + (size_t)b * SEQ * DIMV;

  half16 qf[16];
#pragma unroll
  for (int kk = 0; kk < 16; ++kk)
    qf[kk] = load_afrag(Q + (size_t)m0 * DIMV + kk * 32, DIMV);

  const f32x8 zero8 = {0.f,0.f,0.f,0.f,0.f,0.f,0.f,0.f};
  f32x8 Ob[32];
#pragma unroll
  for (int t = 0; t < 32; ++t) Ob[t] = zero8;
  float mst[8], lst[8];
#pragma unroll
  for (int v = 0; v < 8; ++v) { mst[v] = -__builtin_inff(); lst[v] = 0.f; }

  // stage 0 DMA: K tile rows = keys (32 x 512), V tile rows = dims (512 x 32)
  if (wv == 0) {
    tdm_load_2d(lds_off(&Kb[0][0]), K,  DIMV, SEQ,  DIMV, KT,  DIMV, 7, 3, true);
    tdm_load_2d(lds_off(&Vb[0][0]), Vt, SEQ,  DIMV, KT,  DIMV, SEQ,  3, 3, true);
  }

  _Float16* pb = &pbuf[wv][0];
  int rbase = (lane >> 4) << 3;
  int cidx  = lane & 15;
  const int NIT = SEQ / KT;

  for (int it = 0; it < NIT; ++it) {
    int buf = it & 1;
    if (wv == 0) {
      if (it + 1 < NIT) {
        int n1 = (it + 1) * KT;
        tdm_load_2d(lds_off(&Kb[buf ^ 1][0]), K + (size_t)n1 * DIMV,
                    DIMV, SEQ - n1, DIMV, KT, DIMV, 7, 3, true);
        tdm_load_2d(lds_off(&Vb[buf ^ 1][0]), Vt + n1,
                    SEQ - n1, DIMV, KT, DIMV, SEQ, 3, 3, true);
        __builtin_amdgcn_s_wait_tensorcnt(2);   // stage `it` complete
      } else {
        __builtin_amdgcn_s_wait_tensorcnt(0);
      }
    }
    __syncthreads();                            // stage data visible to all waves

    // ---- scores: S(16x32) = Q(16x512) * K^T, 32 WMMAs from LDS frags ----
    f32x8 sacc[2];
    sacc[0] = zero8; sacc[1] = zero8;
#pragma unroll
    for (int kk = 0; kk < 16; ++kk) {
#pragma unroll
      for (int j = 0; j < 2; ++j) {
        half16 bf = load_bfrag(&Kb[buf][(size_t)(j*16) * KSTRIDE + kk*32], KSTRIDE);
        sacc[j] = wmma16x16x32(qf[kk], bf, sacc[j]);
      }
    }
    // ---- online softmax (row stats replicated per 16-lane half) ----
    float alpha[8];
#pragma unroll
    for (int v = 0; v < 8; ++v) {
      float c = fmaxf(sacc[0][v], sacc[1][v]);
      c = fmaxf(c, __shfl_xor(c, 1));
      c = fmaxf(c, __shfl_xor(c, 2));
      c = fmaxf(c, __shfl_xor(c, 4));
      c = fmaxf(c, __shfl_xor(c, 8));
      float mnew = fmaxf(mst[v], c);
      alpha[v] = __builtin_amdgcn_exp2f(mst[v] - mnew);
      mst[v] = mnew;
    }
#pragma unroll
    for (int v = 0; v < 8; ++v) {
      float rs = 0.f;
#pragma unroll
      for (int j = 0; j < 2; ++j) {
        float e = __builtin_amdgcn_exp2f(sacc[j][v] - mst[v]);
        sacc[j][v] = e;
        rs += e;
      }
      rs += __shfl_xor(rs, 1);
      rs += __shfl_xor(rs, 2);
      rs += __shfl_xor(rs, 4);
      rs += __shfl_xor(rs, 8);
      lst[v] = lst[v]*alpha[v] + rs;
    }
    // ---- P -> padded per-wave LDS, re-fragment as A ----
#pragma unroll
    for (int j = 0; j < 2; ++j)
#pragma unroll
      for (int v = 0; v < 8; ++v)
        pb[(rbase + v)*PSTRIDE + j*16 + cidx] = (_Float16)sacc[j][v];
#pragma unroll
    for (int t = 0; t < 32; ++t)
#pragma unroll
      for (int v = 0; v < 8; ++v)
        Ob[t][v] *= alpha[v];
    half16 pa = load_afrag(pb, PSTRIDE);
    // ---- O += P(16x32) * V(32x512): 32 WMMAs from LDS frags ----
#pragma unroll
    for (int t = 0; t < 32; ++t) {
      half16 vb = load_bfrag(&Vb[buf][(size_t)(t*16) * VSTRIDE], VSTRIDE);
      Ob[t] = wmma16x16x32(pa, vb, Ob[t]);
    }
    __syncthreads();                            // all reads done before overwrite
  }
  // ---- finalize ----
  float rl[8];
#pragma unroll
  for (int v = 0; v < 8; ++v) rl[v] = __builtin_amdgcn_rcpf(lst[v]);
#pragma unroll
  for (int t = 0; t < 32; ++t) {
#pragma unroll
    for (int v = 0; v < 8; ++v) {
      X[(size_t)(m0 + rbase + v) * DIMV + t*16 + cidx] = (_Float16)(Ob[t][v] * rl[v]);
    }
  }
}

// ---------------------------------------------------------------------------
// Kernel 4: O-projection GEMM into LDS, then dual LayerNorm.
// ---------------------------------------------------------------------------
__global__ __launch_bounds__(256) void oproj_ln_kernel(
    const _Float16* __restrict__ XH, const _Float16* __restrict__ WoH,
    const float* __restrict__ bo, const float* __restrict__ gt,
    const float* __restrict__ bt, const float* __restrict__ gi,
    const float* __restrict__ bi, float* __restrict__ out) {
  __shared__ float xs[16][DIMV];
  int lane = threadIdx.x & 31;
  int wv = threadIdx.x >> 5;
  int m0 = blockIdx.x * 16;
  int rbase = (lane >> 4) << 3;
  int cidx  = lane & 15;

  half16 af[16];
#pragma unroll
  for (int kk = 0; kk < 16; ++kk)
    af[kk] = load_afrag(XH + (size_t)m0 * DIMV + kk*32, DIMV);

  const f32x8 zero8 = {0.f,0.f,0.f,0.f,0.f,0.f,0.f,0.f};
#pragma unroll
  for (int q = 0; q < 4; ++q) {
    int n0 = (wv*4 + q) * 16;
    f32x8 acc = zero8;
#pragma unroll
    for (int kk = 0; kk < 16; ++kk) {
      half16 bf = load_bfrag(WoH + (size_t)n0 * DIMV + kk*32, DIMV);
      acc = wmma16x16x32(af[kk], bf, acc);
    }
    float bv = bo[n0 + cidx];
#pragma unroll
    for (int v = 0; v < 8; ++v)
      xs[rbase + v][n0 + cidx] = acc[v] + bv;
  }
  __syncthreads();
#pragma unroll
  for (int rr = 0; rr < 2; ++rr) {
    int r = wv*2 + rr;
    float x[16];
#pragma unroll
    for (int j = 0; j < 16; ++j) x[j] = xs[r][lane*16 + j];
    float sum = 0.f, sq = 0.f;
#pragma unroll
    for (int j = 0; j < 16; ++j) { sum += x[j]; sq += x[j]*x[j]; }
#pragma unroll
    for (int off = 1; off < 32; off <<= 1) {
      sum += __shfl_xor(sum, off);
      sq  += __shfl_xor(sq, off);
    }
    float mu = sum * (1.f/DIMV);
    float var = sq * (1.f/DIMV) - mu*mu;
    float rstd = __builtin_amdgcn_rsqf(var + 1e-5f);
    size_t m = (size_t)m0 + r;
    size_t o1 = m * DIMV;
    size_t o2 = (size_t)NB * SEQ * DIMV + m * DIMV;
#pragma unroll
    for (int j = 0; j < 16; ++j) {
      int d = lane*16 + j;
      float xn = (x[j]-mu)*rstd;
      out[o1 + d] = xn * gt[d] + bt[d];
      out[o2 + d] = xn * gi[d] + bi[d];
    }
  }
}

// ---------------------------------------------------------------------------
extern "C" void kernel_launch(void* const* d_in, const int* in_sizes, int n_in,
                              void* d_out, int out_size, void* d_ws, size_t ws_size,
                              hipStream_t stream) {
  (void)in_sizes; (void)n_in; (void)out_size; (void)ws_size;
  const float* text = (const float*)d_in[0];
  const float* img  = (const float*)d_in[1];
  const float* gt   = (const float*)d_in[2];
  const float* bt   = (const float*)d_in[3];
  const float* gi   = (const float*)d_in[4];
  const float* bi   = (const float*)d_in[5];
  const float* Wq   = (const float*)d_in[6];
  const float* bq   = (const float*)d_in[7];
  const float* Wkt  = (const float*)d_in[8];
  const float* bkt  = (const float*)d_in[9];
  const float* Wvt  = (const float*)d_in[10];
  const float* bvt  = (const float*)d_in[11];
  const float* Wki  = (const float*)d_in[12];
  const float* bki  = (const float*)d_in[13];
  const float* Wvi  = (const float*)d_in[14];
  const float* bvi  = (const float*)d_in[15];
  const float* Wo   = (const float*)d_in[16];
  const float* bo   = (const float*)d_in[17];
  float* out = (float*)d_out;

  const size_t MATE = (size_t)NB * SEQ * DIMV;   // 8,388,608 f16 elements
  _Float16* normsH = (_Float16*)d_ws;
  _Float16* QH  = normsH + MATE;
  _Float16* KH  = QH  + MATE;
  _Float16* VtH = KH  + MATE;
  _Float16* XH  = VtH + MATE;
  _Float16* WH  = XH  + MATE;                    // 6 * 262144 elements
  float* bqS = (float*)(WH + (size_t)6*262144);

  const float qscale = 0.044194173824159216f * 1.4426950408889634f;

  wcvt_kernel<<<6144, 256, 0, stream>>>(Wq, Wkt, Wvt, Wki, Wvi, Wo, bq, WH, bqS, qscale);
  ln_in_kernel<<<2048, 256, 0, stream>>>(text, img, gt, bt, gi, bi, normsH);
  proj_kernel<<<12288, 256, 0, stream>>>(normsH, WH, bqS, bkt, bki, bvt, bvi, QH, KH, VtH);
  fa_kernel<<<128, 256, 0, stream>>>(QH, KH, VtH, XH);
  oproj_ln_kernel<<<1024, 256, 0, stream>>>(XH, WH + (size_t)5*262144, bo, gt, bt, gi, bi, out);
}
